// MultiHeadAttentionPart_86723979641319
// MI455X (gfx1250) — compile-verified, matrix operand order checked by reading, not run on hardware
//
#include <hip/hip_runtime.h>

// ---------------------------------------------------------------------------
// MHA forward for MI455X (gfx1250): bf16 WMMA everywhere.
// Shapes fixed by the reference: B=4, S=2048, E=1024, H=16, D=64.
// ---------------------------------------------------------------------------

typedef __bf16 v16bf __attribute__((ext_vector_type(16)));
typedef float  v8f   __attribute__((ext_vector_type(8)));

#define WMMA_BF16(a, b, c) \
  __builtin_amdgcn_wmma_f32_16x16x32_bf16(false, (a), false, (b), (short)0, (c), false, false)

constexpr int Bb = 4, Ss = 2048, Ee = 1024, Hh = 16, Dd = 64;

// float -> bf16, round-to-nearest-even (pure bit manipulation)
__device__ __forceinline__ __bf16 f2bf(float f) {
  unsigned u = __builtin_bit_cast(unsigned, f);
  unsigned r = (u + 0x7FFFu + ((u >> 16) & 1u)) >> 16;
  unsigned short h = (unsigned short)r;
  return __builtin_bit_cast(__bf16, h);
}

// 16-bit A/B fragment loader (ISA 7.12.2): element e holds
//   k = (e & 7) + (e >> 3)*16 + half*8
// => two contiguous 16B chunks per lane (fuses to ds_load_b128 pairs).
__device__ __forceinline__ v16bf ld_frag(const __bf16* base, int half) {
  v16bf f;
  const __bf16* p = base + half * 8;
#pragma unroll
  for (int e = 0; e < 8; ++e) {
    f[e]     = p[e];
    f[e + 8] = p[e + 16];
  }
  return f;
}

// Async global -> LDS 16-byte copy (CDNA5 ASYNCcnt path). Generic LDS pointers
// carry the LDS byte offset in addr[31:0] (ISA 10.2 aperture mapping).
__device__ __forceinline__ void async_copy16(const __bf16* gsrc, __bf16* ldst) {
  const unsigned lds = (unsigned)(uintptr_t)ldst;
  asm volatile("global_load_async_to_lds_b128 %0, %1, off"
               :: "v"(lds), "v"((unsigned long long)(uintptr_t)gsrc)
               : "memory");
}
__device__ __forceinline__ void wait_async0() {
  asm volatile("s_wait_asynccnt 0" ::: "memory");
}

// ---------------------------------------------------------------------------
// GEMM: C[M,N] = X[M,K] * W[N,K]^T   (M=8192, N=K=1024)
// Block: 256 thr = 8 waves; block tile 128x128; wave tile 32x64; K-step 32.
// Double-buffered LDS, global->reg->LDS pipeline (fp32->bf16 in regs).
// ---------------------------------------------------------------------------
template <bool IN_BF16, bool OUT_F32_BIAS, bool OUT_BHSD>
__global__ __launch_bounds__(256)
void gemm_xwT(const void* __restrict__ Xv, const float* __restrict__ W,
              void* __restrict__ Out, const float* __restrict__ bias) {
  constexpr int N = Ee, K = Ee;
  constexpr int BM = 128, BN = 128, BK = 32, LDT = BK + 8;  // pad: conflict-free
  constexpr int KITER = K / BK;  // 32
  __shared__ alignas(16) __bf16 As[2][BM * LDT];
  __shared__ alignas(16) __bf16 Bs[2][BN * LDT];

  const int tid  = threadIdx.x;
  const int lane = tid & 31, wid = tid >> 5;
  const int half = lane >> 4, l16 = lane & 15;
  const int wm = wid & 3, wn = wid >> 2;                 // 4x2 wave grid
  const int m0 = blockIdx.x * BM, n0 = blockIdx.y * BN;

  const float*  Xf = (const float*)Xv;
  const __bf16* Xb = (const __bf16*)Xv;

  struct alignas(8) bf4 { __bf16 x, y, z, w; };
  bf4 ar[4], br[4];

  auto g2r = [&](int k0) {
#pragma unroll
    for (int t = 0; t < 4; ++t) {
      const int i = tid + t * 256;
      const int r = i >> 3, c = (i & 7) * 4;
      if constexpr (IN_BF16) {
        ar[t] = *(const bf4*)&Xb[(size_t)(m0 + r) * K + k0 + c];
      } else {
        const float4 v = *(const float4*)&Xf[(size_t)(m0 + r) * K + k0 + c];
        ar[t] = { f2bf(v.x), f2bf(v.y), f2bf(v.z), f2bf(v.w) };
      }
      const float4 w = *(const float4*)&W[(size_t)(n0 + r) * K + k0 + c];
      br[t] = { f2bf(w.x), f2bf(w.y), f2bf(w.z), f2bf(w.w) };
    }
  };
  auto r2s = [&](int buf) {
#pragma unroll
    for (int t = 0; t < 4; ++t) {
      const int i = tid + t * 256;
      const int r = i >> 3, c = (i & 7) * 4;
      *(bf4*)&As[buf][r * LDT + c] = ar[t];
      *(bf4*)&Bs[buf][r * LDT + c] = br[t];
    }
  };

  v8f acc[2][4] = {};

  g2r(0);
  r2s(0);
  __syncthreads();

  for (int kk = 0; kk < KITER; ++kk) {
    const int cur = kk & 1;
    if (kk + 1 < KITER) g2r((kk + 1) * BK);  // overlap next stage with WMMA

    const __bf16* Ab = &As[cur][0];
    const __bf16* Bp = &Bs[cur][0];
    v16bf a0 = ld_frag(&Ab[(wm * 32 + l16) * LDT], half);
    v16bf a1 = ld_frag(&Ab[(wm * 32 + 16 + l16) * LDT], half);
#pragma unroll
    for (int nn = 0; nn < 4; ++nn) {
      v16bf bfr = ld_frag(&Bp[(wn * 64 + nn * 16 + l16) * LDT], half);
      acc[0][nn] = WMMA_BF16(a0, bfr, acc[0][nn]);
      acc[1][nn] = WMMA_BF16(a1, bfr, acc[1][nn]);
    }

    if (kk + 1 < KITER) r2s((kk + 1) & 1);
    __syncthreads();
  }

  // store: C layout lane(l16)=n, half gives +8 rows, VGPR r gives +r rows
#pragma unroll
  for (int mm = 0; mm < 2; ++mm)
#pragma unroll
    for (int nn = 0; nn < 4; ++nn) {
      const int mbase = m0 + wm * 32 + mm * 16 + half * 8;
      const int n = n0 + wn * 64 + nn * 16 + l16;
#pragma unroll
      for (int r = 0; r < 8; ++r) {
        const int m = mbase + r;
        const float vv = acc[mm][nn][r];
        if constexpr (OUT_F32_BIAS) {
          ((float*)Out)[(size_t)m * N + n] = vv + bias[n];
        } else if constexpr (OUT_BHSD) {
          const int bb = m >> 11, s = m & (Ss - 1);
          const int hh = n >> 6, dd = n & (Dd - 1);
          ((__bf16*)Out)[(((size_t)(bb * Hh + hh)) * Ss + s) * Dd + dd] = f2bf(vv);
        } else {
          ((__bf16*)Out)[(size_t)m * N + n] = f2bf(vv);
        }
      }
    }
}

// ---------------------------------------------------------------------------
// Flash attention: Q,K,V bf16 in [B,H,S,D]; O bf16 in [B,S,H,D] (= [B*S, E]).
// 8 waves x 32 queries per block; 64-key chunks. K staged with CDNA5 async
// global->LDS loads; V staged transposed; P re-staged per-wave for PV WMMA.
// ---------------------------------------------------------------------------
__global__ __launch_bounds__(256)
void flash_attn(const __bf16* __restrict__ Q, const __bf16* __restrict__ Km,
                const __bf16* __restrict__ V, const int* __restrict__ mask,
                __bf16* __restrict__ O) {
  constexpr int CH = 64;           // keys per chunk
  constexpr int LDK = Dd + 8;      // 72: K tile [key][d]
  constexpr int LDV = CH + 8;      // 72: V^T tile [d][key]
  constexpr int LDP = CH + 8;      // 72: per-wave P tile [q][key]
  __shared__ alignas(16) __bf16 Ks[CH * LDK];
  __shared__ alignas(16) __bf16 Vt[Dd * LDV];
  __shared__ alignas(16) __bf16 Ps[8 * 32 * LDP];

  const int tid  = threadIdx.x;
  const int lane = tid & 31, wid = tid >> 5;
  const int half = lane >> 4, l16 = lane & 15;
  const int bh = blockIdx.y, b = bh >> 4, h = bh & (Hh - 1);
  const int s0 = blockIdx.x * 256 + wid * 32;

  const size_t headoff = (size_t)bh * Ss * Dd;
  const __bf16* Qh = Q + headoff;
  const __bf16* Kh = Km + headoff;
  const __bf16* Vh = V + headoff;
  const int* mrow = mask + (size_t)b * Ss;
  __bf16* Pw = &Ps[wid * 32 * LDP];

  // Q fragments [qt][ks]: loaded once, rows contiguous along D
  v16bf qf[2][2];
#pragma unroll
  for (int qt = 0; qt < 2; ++qt) {
    const __bf16* qrow = Qh + (size_t)(s0 + qt * 16 + l16) * Dd;
    qf[qt][0] = ld_frag(qrow, half);
    qf[qt][1] = ld_frag(qrow + 32, half);
  }

  v8f oacc[2][4] = {};
  float mrun[2][8], lrun[2][8];
#pragma unroll
  for (int qt = 0; qt < 2; ++qt)
#pragma unroll
    for (int r = 0; r < 8; ++r) { mrun[qt][r] = -3.0e38f; lrun[qt][r] = 0.0f; }

  const float scale = 0.125f;  // 1/sqrt(D_HEAD=64)

  for (int kc = 0; kc < Ss; kc += CH) {
    // stage K chunk [key][d]: 512 x 16B async global->LDS copies
#pragma unroll
    for (int t = 0; t < 2; ++t) {
      const int j = tid + t * 256;
      const int key = j >> 3, c8 = (j & 7) * 8;
      async_copy16(&Kh[(size_t)(kc + key) * Dd + c8], &Ks[key * LDK + c8]);
    }
    // stage V chunk transposed: Vt[d][key]
#pragma unroll
    for (int i = tid; i < CH * Dd / 4; i += 256) {
      const int r = i >> 4, c = (i & 15) * 4;  // r=key, c=d
      const __bf16* s = &Vh[(size_t)(kc + r) * Dd + c];
      Vt[(c + 0) * LDV + r] = s[0];
      Vt[(c + 1) * LDV + r] = s[1];
      Vt[(c + 2) * LDV + r] = s[2];
      Vt[(c + 3) * LDV + r] = s[3];
    }
    wait_async0();
    __syncthreads();

#pragma unroll
    for (int qt = 0; qt < 2; ++qt) {
      // scores: four 16-key column tiles, K-dim = D = 2 x 32
      v8f sc[4] = {};
#pragma unroll
      for (int kt = 0; kt < 4; ++kt)
#pragma unroll
        for (int ks = 0; ks < 2; ++ks) {
          v16bf bk = ld_frag(&Ks[(kt * 16 + l16) * LDK + ks * 32], half);
          sc[kt] = WMMA_BF16(qf[qt][ks], bk, sc[kt]);
        }

      // mask + scale (lane -> key column; same mask for all rows r)
      int mv[4];
#pragma unroll
      for (int kt = 0; kt < 4; ++kt) mv[kt] = mrow[kc + kt * 16 + l16];
      float p[4][8];
#pragma unroll
      for (int kt = 0; kt < 4; ++kt)
#pragma unroll
        for (int r = 0; r < 8; ++r)
          p[kt][r] = mv[kt] ? sc[kt][r] * scale : -1.0e10f;

      // online softmax per row (row lives across a 16-lane group)
#pragma unroll
      for (int r = 0; r < 8; ++r) {
        float v = fmaxf(fmaxf(p[0][r], p[1][r]), fmaxf(p[2][r], p[3][r]));
        v = fmaxf(v, __shfl_xor(v, 1));
        v = fmaxf(v, __shfl_xor(v, 2));
        v = fmaxf(v, __shfl_xor(v, 4));
        v = fmaxf(v, __shfl_xor(v, 8));
        const float mnew  = fmaxf(mrun[qt][r], v);
        const float alpha = __expf(mrun[qt][r] - mnew);
        mrun[qt][r] = mnew;
        float rs = 0.0f;
#pragma unroll
        for (int kt = 0; kt < 4; ++kt) {
          p[kt][r] = __expf(p[kt][r] - mnew);
          rs += p[kt][r];
        }
        rs += __shfl_xor(rs, 1);
        rs += __shfl_xor(rs, 2);
        rs += __shfl_xor(rs, 4);
        rs += __shfl_xor(rs, 8);
        lrun[qt][r] = lrun[qt][r] * alpha + rs;
#pragma unroll
        for (int nt = 0; nt < 4; ++nt) oacc[qt][nt][r] *= alpha;
      }

      // C-layout -> A-layout re-stage of P through per-wave LDS
#pragma unroll
      for (int kt = 0; kt < 4; ++kt)
#pragma unroll
        for (int r = 0; r < 8; ++r)
          Pw[(qt * 16 + half * 8 + r) * LDP + kt * 16 + l16] = f2bf(p[kt][r]);
    }

    // O += P (32x64) x V (64x64): 2 K-steps x 4 d-column tiles per qt
#pragma unroll
    for (int qt = 0; qt < 2; ++qt) {
      v16bf pa0 = ld_frag(&Pw[(qt * 16 + l16) * LDP], half);
      v16bf pa1 = ld_frag(&Pw[(qt * 16 + l16) * LDP + 32], half);
#pragma unroll
      for (int nt = 0; nt < 4; ++nt) {
        v16bf bv0 = ld_frag(&Vt[(nt * 16 + l16) * LDV], half);
        v16bf bv1 = ld_frag(&Vt[(nt * 16 + l16) * LDV + 32], half);
        oacc[qt][nt] = WMMA_BF16(pa0, bv0, oacc[qt][nt]);
        oacc[qt][nt] = WMMA_BF16(pa1, bv1, oacc[qt][nt]);
      }
    }
    __syncthreads();
  }

  // epilogue: normalize and store O as [B,S,H,D] bf16
#pragma unroll
  for (int qt = 0; qt < 2; ++qt)
#pragma unroll
    for (int r = 0; r < 8; ++r) {
      const int s = s0 + qt * 16 + half * 8 + r;
      const float inv = 1.0f / lrun[qt][r];
#pragma unroll
      for (int nt = 0; nt < 4; ++nt) {
        const int d = nt * 16 + l16;
        O[(((size_t)b * Ss + s) * Hh + h) * Dd + d] = f2bf(oacc[qt][nt][r] * inv);
      }
    }
}

// ---------------------------------------------------------------------------
extern "C" void kernel_launch(void* const* d_in, const int* in_sizes, int n_in,
                              void* d_out, int out_size, void* d_ws, size_t ws_size,
                              hipStream_t stream) {
  (void)in_sizes; (void)n_in; (void)out_size; (void)ws_size;
  const float* q    = (const float*)d_in[0];
  const float* k    = (const float*)d_in[1];
  const float* v    = (const float*)d_in[2];
  const int*   mask = (const int*)d_in[3];
  const float* Wq   = (const float*)d_in[4];
  const float* Wk   = (const float*)d_in[5];
  const float* Wv   = (const float*)d_in[6];
  const float* Wo   = (const float*)d_in[7];
  const float* bo   = (const float*)d_in[8];

  const size_t nper = (size_t)Bb * Hh * Ss * Dd;  // 8.39M elems (bf16)
  __bf16* Qb = (__bf16*)d_ws;
  __bf16* Kb = Qb + nper;
  __bf16* Vb = Kb + nper;
  __bf16* Ob = Vb + nper;   // total ws use: 4 * 16.8 MB = 67 MB

  dim3 blk(256);
  dim3 gGemm(64, 8);  // (8192/128, 1024/128)

  gemm_xwT<false, false, true><<<gGemm, blk, 0, stream>>>(q, Wq, Qb, nullptr);
  gemm_xwT<false, false, true><<<gGemm, blk, 0, stream>>>(k, Wk, Kb, nullptr);
  gemm_xwT<false, false, true><<<gGemm, blk, 0, stream>>>(v, Wv, Vb, nullptr);

  dim3 gAttn(Ss / 256, Bb * Hh);  // (8, 64)
  flash_attn<<<gAttn, blk, 0, stream>>>(Qb, Kb, Vb, mask, Ob);

  gemm_xwT<true, true, false><<<gGemm, blk, 0, stream>>>(Ob, Wo, d_out, bo);
}
